// MyModel_81595788689658
// MI455X (gfx1250) — matrix operand order repaired and checked by reading, not compiled, not run on hardware
//
#include <hip/hip_runtime.h>
#include <stdint.h>

#define D_IN   2048
#define D_HID  2048
#define D_OUT  300000
#define BATCH  256

typedef __attribute__((ext_vector_type(16))) __bf16 v16bf;
typedef __attribute__((ext_vector_type(8)))  float  v8f;
typedef __attribute__((ext_vector_type(4)))  uint32_t u32x4;
typedef __attribute__((ext_vector_type(8)))  int32_t  i32x8;
typedef __attribute__((ext_vector_type(4)))  int32_t  i32x4;

union V16U {
    v16bf    v;
    uint16_t h[16];
    uint4    q[2];
};

// Split fp32 into bf16 hi (truncated top16) + bf16 lo (residual). x - hi is
// exact (same exponent, Sterbenz), so hi+lo carries ~17 significand bits.
__device__ __forceinline__ void split2(float x, uint16_t& hi, uint16_t& lo) {
    uint32_t xb = __float_as_uint(x);
    float hf = __uint_as_float(xb & 0xFFFF0000u);
    float r  = x - hf;
    hi = (uint16_t)(xb >> 16);
    lo = (uint16_t)(__float_as_uint(r) >> 16);
}

__device__ __forceinline__ v8f wmma_bf16(v16bf a, v16bf b, v8f c) {
    return __builtin_amdgcn_wmma_f32_16x16x32_bf16(
        false, a, false, b, (short)0, c, false, false);
}

// ---------------- kernel 0: split x into bf16 hi/lo planes ----------------
__global__ __launch_bounds__(256) void k_split(const float* __restrict__ src,
                                               uint16_t* __restrict__ hi,
                                               uint16_t* __restrict__ lo,
                                               int n) {
    int i = blockIdx.x * 256 + threadIdx.x;
    if (i < n) split2(src[i], hi[i], lo[i]);
}

// ---------------- kernel 1: h = x @ W1^T + b1 (split-bf16 WMMA) -----------
__global__ __launch_bounds__(32) void k_l1(const uint16_t* __restrict__ xHi,
                                           const uint16_t* __restrict__ xLo,
                                           const float* __restrict__ w1,
                                           const float* __restrict__ b1,
                                           uint16_t* __restrict__ hHi,
                                           uint16_t* __restrict__ hLo) {
    const int lane  = threadIdx.x & 31;
    const int rl    = lane & 15;
    const int hs    = (lane >> 4) & 1;
    const int koffA = hs * 16;   // A: lanes 0-15 -> K 0..15; 16-31 -> K 16..31
    const int koffB = hs * 16;   // B: lanes 0-15 -> K 0..15; 16-31 -> K 16..31
    const int ntile = blockIdx.x;
    const int mbase = blockIdx.y * 4;
    const int col   = ntile * 16 + rl;

    v8f zero = {0.f,0.f,0.f,0.f,0.f,0.f,0.f,0.f};
    v8f acc[4];
    acc[0] = zero; acc[1] = zero; acc[2] = zero; acc[3] = zero;

    for (int k0 = 0; k0 < D_IN; k0 += 32) {
        union { float4 v4[4]; float f[16]; } wb;
        const float4* wp = (const float4*)(w1 + (size_t)col * D_IN + k0 + koffB);
        wb.v4[0] = wp[0]; wb.v4[1] = wp[1]; wb.v4[2] = wp[2]; wb.v4[3] = wp[3];
        V16U bhi, blo;
        #pragma unroll
        for (int i = 0; i < 16; ++i) split2(wb.f[i], bhi.h[i], blo.h[i]);

        #pragma unroll
        for (int mt = 0; mt < 4; ++mt) {
            const int row = (mbase + mt) * 16 + rl;
            const uint16_t* ph = xHi + (size_t)row * D_IN + k0 + koffA;
            const uint16_t* pl = xLo + (size_t)row * D_IN + k0 + koffA;
            V16U ahi, alo;
            ahi.q[0] = *(const uint4*)(ph);
            ahi.q[1] = *(const uint4*)(ph + 8);
            alo.q[0] = *(const uint4*)(pl);
            alo.q[1] = *(const uint4*)(pl + 8);
            acc[mt] = wmma_bf16(ahi.v, bhi.v, acc[mt]);
            acc[mt] = wmma_bf16(ahi.v, blo.v, acc[mt]);
            acc[mt] = wmma_bf16(alo.v, bhi.v, acc[mt]);
        }
    }

    const float bias = b1[col];
    #pragma unroll
    for (int mt = 0; mt < 4; ++mt) {
        const int rowBase = (mbase + mt) * 16 + hs * 8;
        #pragma unroll
        for (int r = 0; r < 8; ++r) {
            float v = acc[mt][r] + bias;
            uint16_t hb, lb;
            split2(v, hb, lb);
            hHi[(size_t)(rowBase + r) * D_HID + col] = hb;
            hLo[(size_t)(rowBase + r) * D_HID + col] = lb;
        }
    }
}

// ---------------- kernel 2: out = h @ W2^T + b2 (dominant GEMM) -----------
// Block = 256 threads / 8 waves. Wave = 2 N-tiles x 8 M-tiles (M split across
// wave pairs) -> block covers 256 rows x 128 cols, so W2 streams from HBM
// exactly once. h hi/lo slab (32 K x 256 rows x 2 planes) is DMA'd into LDS
// by the Tensor Data Mover, double buffered, with hardware padding giving an
// 80B row pitch (20-bank stride -> conflict-free ds_load_b128 for 16 lanes).
#define L2_PITCH 40           // uint16 elems per padded row (64B data + 16B pad)
#define NSTEPS   (D_HID / 32)

// One TDM op moves both planes: 3D tile x=32 elems, y=256 rows (stride 2048),
// z=2 planes (stride BATCH*D_HID). LDS pad: 4 DWORDs after every 16 DWORDs.
__device__ __forceinline__ void tdm_slab_load(const uint16_t* gsrc, uint32_t lds_off) {
    uint64_t ga = (uint64_t)(uintptr_t)gsrc;
    u32x4 g0;
    g0[0] = 1u;                                   // count=1, user mode
    g0[1] = lds_off;                              // LDS byte address
    g0[2] = (uint32_t)ga;                         // global_addr[31:0]
    g0[3] = (uint32_t)(ga >> 32) | (2u << 30);    // global_addr[56:32] | type=2
    i32x8 g1;
    g1[0] = (int)((1u << 16)      // data_size = 2 bytes
                | (1u << 20)      // pad_enable
                | (3u << 22)      // pad_interval: 16 DWORDs (64B)
                | (3u << 25));    // pad_amount:   4 DWORDs (16B)
    g1[1] = (int)(2048u << 16);           // tensor_dim0 = 2048 (bits 79:48, low16)
    g1[2] = (int)(256u  << 16);           // tensor_dim1 = 256  (bits 111:80, low16)
    g1[3] = (int)(32u   << 16);           // tile_dim0 = 32 (bits 127:112)
    g1[4] = (int)(256u | (2u << 16));     // tile_dim1 = 256 | tile_dim2 = 2
    g1[5] = (int)2048;                    // tensor_dim0_stride = 2048 (row stride)
    g1[6] = 0;                            // dim0_stride hi | dim1_stride low16 (524288 & 0xFFFF = 0)
    g1[7] = (int)((uint32_t)((size_t)BATCH * D_HID >> 16));  // dim1_stride hi = 8
    i32x4 g2; g2[0] = 2; g2[1] = 0; g2[2] = 0; g2[3] = 0;    // tensor_dim2 = 2
    i32x4 g3; g3[0] = 0; g3[1] = 0; g3[2] = 0; g3[3] = 0;
    i32x8 g4 = {0, 0, 0, 0, 0, 0, 0, 0};                     // unused on this path
    // clang-23 / therock-10.0 lane: 6-arg form (u32x4, i32x8, i32x4, i32x4, i32x8, i32)
    __builtin_amdgcn_tensor_load_to_lds(g0, g1, g2, g3, g4, 0);
}

__global__ __launch_bounds__(256) void k_l2(const uint16_t* __restrict__ hHi,
                                            const float* __restrict__ w2,
                                            const float* __restrict__ b2,
                                            float* __restrict__ out) {
    // [dbuf][plane][row][pitch]; planes contiguous so one TDM fills both.
    __shared__ __attribute__((aligned(16))) uint16_t lds[2][2][BATCH][L2_PITCH];

    const int tid   = threadIdx.x;
    const int lane  = tid & 31;
    const int w     = tid >> 5;
    const int pair  = w >> 1;
    const int mhalf = w & 1;
    const int rl    = lane & 15;
    const int hs    = (lane >> 4) & 1;
    const int koffA = hs * 16;
    const int koffB = hs * 16;
    const int mrow0 = mhalf * 128;

    int  n0[2];
    bool valid[2];
    #pragma unroll
    for (int nt = 0; nt < 2; ++nt) {
        n0[nt]    = (blockIdx.x * 8 + pair * 2 + nt) * 16;
        valid[nt] = (n0[nt] < D_OUT);
    }

    v8f zero = {0.f,0.f,0.f,0.f,0.f,0.f,0.f,0.f};
    v8f acc[2][8];
    #pragma unroll
    for (int nt = 0; nt < 2; ++nt)
        #pragma unroll
        for (int mt = 0; mt < 8; ++mt) acc[nt][mt] = zero;

    // --- preheader: TDM prefetch slab 0; load + convert B for step 0 -------
    if (w == 0) {
        tdm_slab_load(hHi, (uint32_t)(uintptr_t)&lds[0][0][0][0]);
    }

    float4 braw[2][4];
    #pragma unroll
    for (int nt = 0; nt < 2; ++nt) {
        if (valid[nt]) {
            const float4* wp = (const float4*)(w2 + (size_t)(n0[nt] + rl) * D_HID + koffB);
            braw[nt][0] = wp[0]; braw[nt][1] = wp[1];
            braw[nt][2] = wp[2]; braw[nt][3] = wp[3];
        } else {
            float4 z = {0.f,0.f,0.f,0.f};
            braw[nt][0] = z; braw[nt][1] = z; braw[nt][2] = z; braw[nt][3] = z;
        }
    }
    V16U bhi[2], blo[2];
    #pragma unroll
    for (int nt = 0; nt < 2; ++nt) {
        union { float4 v4[4]; float f[16]; } wb;
        wb.v4[0] = braw[nt][0]; wb.v4[1] = braw[nt][1];
        wb.v4[2] = braw[nt][2]; wb.v4[3] = braw[nt][3];
        #pragma unroll
        for (int i = 0; i < 16; ++i) split2(wb.f[i], bhi[nt].h[i], blo[nt].h[i]);
    }

    for (int s = 0; s < NSTEPS; ++s) {
        const int cb = s & 1;

        // --- wave 0: prefetch slab s+1, then drain slab s (in-order TDM) ---
        if (w == 0) {
            if (s + 1 < NSTEPS) {
                tdm_slab_load(hHi + (size_t)(s + 1) * 32,
                              (uint32_t)(uintptr_t)&lds[(s + 1) & 1][0][0][0]);
                __builtin_amdgcn_s_wait_tensorcnt(1);
            } else {
                __builtin_amdgcn_s_wait_tensorcnt(0);
            }
        }
        __syncthreads();   // publish slab s to all waves

        // --- issue next step's raw B loads (latency hidden behind WMMAs) ---
        const int kn = (s + 1 < NSTEPS) ? (s + 1) * 32 : 0;
        #pragma unroll
        for (int nt = 0; nt < 2; ++nt) {
            if (valid[nt]) {
                const float4* wp =
                    (const float4*)(w2 + (size_t)(n0[nt] + rl) * D_HID + kn + koffB);
                braw[nt][0] = wp[0]; braw[nt][1] = wp[1];
                braw[nt][2] = wp[2]; braw[nt][3] = wp[3];
            }
        }

        // --- 8 M-tiles x (3 split-terms x 2 N-tiles) = 48 WMMAs ------------
        #pragma unroll
        for (int mt = 0; mt < 8; ++mt) {
            const int row = mrow0 + mt * 16 + rl;
            V16U ahi, alo;
            ahi.q[0] = *(const uint4*)&lds[cb][0][row][koffA];
            ahi.q[1] = *(const uint4*)&lds[cb][0][row][koffA + 8];
            alo.q[0] = *(const uint4*)&lds[cb][1][row][koffA];
            alo.q[1] = *(const uint4*)&lds[cb][1][row][koffA + 8];
            // term-outer order: consecutive WMMAs hit different accumulators
            acc[0][mt] = wmma_bf16(ahi.v, bhi[0].v, acc[0][mt]);
            acc[1][mt] = wmma_bf16(ahi.v, bhi[1].v, acc[1][mt]);
            acc[0][mt] = wmma_bf16(ahi.v, blo[0].v, acc[0][mt]);
            acc[1][mt] = wmma_bf16(ahi.v, blo[1].v, acc[1][mt]);
            acc[0][mt] = wmma_bf16(alo.v, bhi[0].v, acc[0][mt]);
            acc[1][mt] = wmma_bf16(alo.v, bhi[1].v, acc[1][mt]);
        }

        // --- convert landed raw B for step s+1 (waits loadcnt here) --------
        #pragma unroll
        for (int nt = 0; nt < 2; ++nt) {
            union { float4 v4[4]; float f[16]; } wb;
            wb.v4[0] = braw[nt][0]; wb.v4[1] = braw[nt][1];
            wb.v4[2] = braw[nt][2]; wb.v4[3] = braw[nt][3];
            #pragma unroll
            for (int i = 0; i < 16; ++i) split2(wb.f[i], bhi[nt].h[i], blo[nt].h[i]);
        }
        __syncthreads();   // all waves done with slab s before TDM overwrites
    }

    #pragma unroll
    for (int nt = 0; nt < 2; ++nt) {
        if (!valid[nt]) continue;
        const int col  = n0[nt] + rl;
        const float bias = b2[col];
        #pragma unroll
        for (int mt = 0; mt < 8; ++mt) {
            const int rowBase = mrow0 + mt * 16 + hs * 8;
            #pragma unroll
            for (int r = 0; r < 8; ++r) {
                out[(size_t)(rowBase + r) * D_OUT + col] = acc[nt][mt][r] + bias;
            }
        }
    }
}

// --------------------------------------------------------------------------
extern "C" void kernel_launch(void* const* d_in, const int* in_sizes, int n_in,
                              void* d_out, int out_size, void* d_ws, size_t ws_size,
                              hipStream_t stream) {
    const float* x  = (const float*)d_in[0];
    const float* w1 = (const float*)d_in[1];
    const float* b1 = (const float*)d_in[2];
    const float* w2 = (const float*)d_in[3];
    const float* b2 = (const float*)d_in[4];
    float* out = (float*)d_out;

    // workspace layout (4 MB): x hi/lo planes, then h hi/lo planes (must be
    // contiguous: k_l2's TDM walks hi->lo as a 3D tensor with plane stride).
    uint16_t* xHi = (uint16_t*)d_ws;
    uint16_t* xLo = xHi + (size_t)BATCH * D_IN;
    uint16_t* hHi = xLo + (size_t)BATCH * D_IN;
    uint16_t* hLo = hHi + (size_t)BATCH * D_HID;

    k_split<<<dim3((BATCH * D_IN + 255) / 256), 256, 0, stream>>>(
        x, xHi, xLo, BATCH * D_IN);

    k_l1<<<dim3(D_HID / 16, BATCH / 64), 32, 0, stream>>>(
        xHi, xLo, w1, b1, hHi, hLo);

    k_l2<<<dim3((D_OUT / 16 + 7) / 8), 256, 0, stream>>>(
        hHi, w2, b2, out);
}
